// Attention_poc_71081708749524
// MI455X (gfx1250) — compile-verified, matrix operand order checked by reading, not run on hardware
//
#include <hip/hip_runtime.h>

// ---------------------------------------------------------------------------
// Problem constants
// ---------------------------------------------------------------------------
#define BATCH   64
#define SEQ     236
#define SEQP    240      // padded to multiple of 16
#define SEQP2   256      // padded to multiple of 32 (WMMA K for P@V)
#define CH      768
#define HEADS   8
#define HDIM    96
#define CQKV    2304     // 3*CH
#define S1B     196
#define S2B     216
#define BIASV   100.0f

typedef __attribute__((ext_vector_type(16))) __bf16   v16bf;
typedef __attribute__((ext_vector_type(8)))  float    v8f;
typedef __attribute__((ext_vector_type(4)))  unsigned int u32x4;
typedef __attribute__((ext_vector_type(4)))  int       v4i;

struct Frag2 { u32x4 lo, hi; };

// Load one 16x32 bf16 A/B fragment (per-lane layout per CDNA5 ISA §7.12.2):
//   lanes 0-15 : row = base+(lane&15), K = 0..7 (bytes 0..15) and 16..23 (bytes 32..47)
//   lanes 16-31: same rows,            K = 8..15 (bytes 16..31) and 24..31 (bytes 48..63)
__device__ __forceinline__ v16bf load_frag(const __bf16* base, int stride,
                                           int row, int kc, int laneHi) {
  const char* p = (const char*)(base + (size_t)row * stride + kc);
  Frag2 f;
  f.lo = *(const u32x4*)(p + (laneHi ? 16 : 0));
  f.hi = *(const u32x4*)(p + 32 + (laneHi ? 16 : 0));
  return __builtin_bit_cast(v16bf, f);
}

__device__ __forceinline__ v8f wmma_bf16(v16bf a, v16bf b, v8f c) {
  return __builtin_amdgcn_wmma_f32_16x16x32_bf16(false, a, false, b,
                                                 (short)0, c, false, false);
}

// Async global->LDS copy of 16 bytes (GLOBAL_LOAD_ASYNC_TO_LDS_B128, ASYNCcnt)
__device__ __forceinline__ void async_copy_b128(const __bf16* g, __bf16* l) {
#if defined(__HIP_DEVICE_COMPILE__)
#if __has_builtin(__builtin_amdgcn_global_load_async_to_lds_b128)
  __builtin_amdgcn_global_load_async_to_lds_b128(
      (__attribute__((address_space(1))) v4i*)(v4i*)g,
      (__attribute__((address_space(3))) v4i*)(v4i*)l, 0, 0);
#else
  unsigned loff = (unsigned)(size_t)(__attribute__((address_space(3))) char*)(char*)l;
  asm volatile("global_load_async_to_lds_b128 %0, %1, off"
               :: "v"(loff), "v"(g) : "memory");
#endif
#else
  (void)g; (void)l;
#endif
}

__device__ __forceinline__ void wait_async0() {
#if defined(__HIP_DEVICE_COMPILE__)
#if __has_builtin(__builtin_amdgcn_s_wait_asynccnt)
  __builtin_amdgcn_s_wait_asynccnt(0);
#else
  asm volatile("s_wait_asynccnt 0x0" ::: "memory");
#endif
#endif
}

// ---------------------------------------------------------------------------
// fp32 -> bf16 converts
// ---------------------------------------------------------------------------
__global__ void cvt_x_kernel(const float* __restrict__ x, __bf16* __restrict__ xh) {
  int idx = blockIdx.x * 256 + threadIdx.x;          // over BATCH*SEQP*CH
  if (idx >= BATCH * SEQP * CH) return;
  int c = idx % CH;
  int r = idx / CH;
  int n = r % SEQP, b = r / SEQP;
  xh[idx] = (n < SEQ) ? (__bf16)x[((size_t)(b * SEQ + n)) * CH + c] : (__bf16)0.0f;
}

__global__ void cvt_w_kernel(const float* __restrict__ w, __bf16* __restrict__ wh, int count) {
  int idx = blockIdx.x * 256 + threadIdx.x;
  if (idx < count) wh[idx] = (__bf16)w[idx];
}

// ---------------------------------------------------------------------------
// Tiled GEMM:  Out[i,j] = sum_k A[i,k] * W[j,k]   (A: MxK bf16, W: NxK bf16)
// Block tile 128x128, K-step 32, 8 waves, each wave 2x4 WMMA tiles.
// Double-buffered LDS filled by async global->LDS copies (overlaps WMMA).
// mode 0: scatter into qkv buffer [3,B,H,SEQP,HDIM] (bf16)
// mode 1: out fp32 [B,SEQ,CH] += bias, dropping pad rows
// ---------------------------------------------------------------------------
#define AS_STRIDE 40   // 32 + 8 pad (keeps 16B alignment, dodges bank conflicts)

__global__ __launch_bounds__(256) void gemm_kernel(
    const __bf16* __restrict__ A, const __bf16* __restrict__ W, int Kdim,
    int mode, __bf16* __restrict__ qkvh, float* __restrict__ out,
    const float* __restrict__ bias) {
  __shared__ __bf16 As[2][128 * AS_STRIDE];
  __shared__ __bf16 Bs[2][128 * AS_STRIDE];

  const int tid    = threadIdx.x;
  const int wave   = tid >> 5;
  const int lane   = tid & 31;
  const int laneHi = lane >> 4;
  const int lrow   = lane & 15;
  const int rowBase = blockIdx.x * 128;
  const int colBase = blockIdx.y * 128;
  const int waveM = wave >> 1;        // 0..3 -> 32 rows each
  const int waveN = wave & 1;         // 0..1 -> 64 cols each
  const int lr  = tid >> 1;           // staging row 0..127
  const int lco = (tid & 1) * 16;     // staging col half (elements)

  v8f acc[2][4] = {};

  auto issue = [&](int buf, int k0) {
    const __bf16* ga = A + (size_t)(rowBase + lr) * Kdim + k0 + lco;
    __bf16* la = &As[buf][lr * AS_STRIDE + lco];
    async_copy_b128(ga, la);
    async_copy_b128(ga + 8, la + 8);
    const __bf16* gb = W + (size_t)(colBase + lr) * Kdim + k0 + lco;
    __bf16* lb = &Bs[buf][lr * AS_STRIDE + lco];
    async_copy_b128(gb, lb);
    async_copy_b128(gb + 8, lb + 8);
  };

  const int nK = Kdim / 32;
  issue(0, 0);
  for (int kt = 0; kt < nK; ++kt) {
    const int cur = kt & 1;
    wait_async0();       // this wave's async fills landed
    __syncthreads();     // => every wave's fills landed
    if (kt + 1 < nK) issue(cur ^ 1, (kt + 1) * 32);   // overlap next fill with math

    v16bf af[2], bfg[4];
    #pragma unroll
    for (int i = 0; i < 2; ++i)
      af[i] = load_frag(&As[cur][0], AS_STRIDE, waveM * 32 + i * 16 + lrow, 0, laneHi);
    #pragma unroll
    for (int j = 0; j < 4; ++j)
      bfg[j] = load_frag(&Bs[cur][0], AS_STRIDE, waveN * 64 + j * 16 + lrow, 0, laneHi);
    #pragma unroll
    for (int i = 0; i < 2; ++i)
      #pragma unroll
      for (int j = 0; j < 4; ++j)
        acc[i][j] = wmma_bf16(af[i], bfg[j], acc[i][j]);
    __syncthreads();     // done reading 'cur' before it is refilled at kt+2
  }

  // epilogue: C/D layout -> element r of VGPR: row = r (+8 for lanes>=16), col = lane&15
  #pragma unroll
  for (int i = 0; i < 2; ++i)
    #pragma unroll
    for (int j = 0; j < 4; ++j)
      #pragma unroll
      for (int r = 0; r < 8; ++r) {
        int rr = laneHi ? (r + 8) : r;
        int gi = rowBase + waveM * 32 + i * 16 + rr;
        int gj = colBase + waveN * 64 + j * 16 + lrow;
        float v = acc[i][j][r];
        int b = gi / SEQP, n = gi % SEQP;
        if (mode == 0) {
          int s = gj / CH, rem = gj % CH;
          int h = rem / HDIM, d = rem % HDIM;
          qkvh[((((size_t)s * BATCH * HEADS) + (size_t)(b * HEADS + h)) * SEQP + n) * HDIM + d] =
              (__bf16)v;
        } else {
          if (n < SEQ)
            out[((size_t)(b * SEQ + n)) * CH + gj] = v + bias[gj];
        }
      }
}

// ---------------------------------------------------------------------------
// Fused attention per (b,h): scores -> block edits -> softmax -> P@V
// ---------------------------------------------------------------------------
#define KS_STRIDE 104   // 96 + 8 pad
#define VT_STRIDE 264   // 256 + 8 pad
#define P_STRIDE  264

#define SMEM_KS   (SEQP * KS_STRIDE * 2)             // 49920
#define SMEM_VT   (HDIM * VT_STRIDE * 2)             // 50688
#define SMEM_SSC  (8 * 16 * SEQP2 * 4)               // 131072
#define SMEM_PS   (8 * 16 * P_STRIDE * 2)            // 67584
#define SMEM_ATTN (SMEM_KS + SMEM_VT + SMEM_SSC + SMEM_PS)

__global__ __launch_bounds__(256) void attn_kernel(
    const __bf16* __restrict__ qkvh, __bf16* __restrict__ ctx) {
  const int bh = blockIdx.x;                 // 0..511
  const int b = bh >> 3, h = bh & 7;
  const int tid    = threadIdx.x;
  const int wave   = tid >> 5;
  const int lane   = tid & 31;
  const int laneHi = lane >> 4;
  const int lrow   = lane & 15;
  const float scale = 0.10206207261596577f;  // 96^-0.5

  extern __shared__ char smem[];
  __bf16* Ks  = (__bf16*)smem;                                   // [SEQP][KS_STRIDE]
  __bf16* Vt  = (__bf16*)(smem + SMEM_KS);                       // [HDIM][VT_STRIDE]
  float*  Ssc = (float*)(smem + SMEM_KS + SMEM_VT);              // [8][16][SEQP2]
  __bf16* Ps  = (__bf16*)(smem + SMEM_KS + SMEM_VT + SMEM_SSC);  // [8][16][P_STRIDE]

  const size_t plane = (size_t)(b * HEADS + h) * (SEQP * HDIM);
  const size_t qkvSz = (size_t)BATCH * HEADS * SEQP * HDIM;
  const __bf16* Qg = qkvh + plane;
  const __bf16* Kg = qkvh + qkvSz + plane;
  const __bf16* Vg = qkvh + 2 * qkvSz + plane;

  // stage K row-major via async copies: 240 rows x 12 b128-chunks (16B aligned both sides)
  for (int i = tid; i < SEQP * (HDIM / 8); i += 256) {
    int n = i / (HDIM / 8), c = i % (HDIM / 8);
    async_copy_b128(Kg + n * HDIM + c * 8, Ks + n * KS_STRIDE + c * 8);
  }
  // stage V transposed (scatter, regular ds stores); zero-pad k in [240,256)
  for (int i = tid; i < SEQP * HDIM; i += 256) {
    int k = i / HDIM, d = i % HDIM;
    Vt[d * VT_STRIDE + k] = Vg[i];
  }
  for (int i = tid; i < HDIM * (SEQP2 - SEQP); i += 256) {
    int d = i / (SEQP2 - SEQP), k = SEQP + i % (SEQP2 - SEQP);
    Vt[d * VT_STRIDE + k] = (__bf16)0.0f;
  }
  wait_async0();
  __syncthreads();

  float*  Sw = Ssc + wave * 16 * SEQP2;
  __bf16* Pw = Ps  + wave * 16 * P_STRIDE;

  for (int rt = wave; rt < SEQP / 16; rt += 8) {     // row tiles
    // Q fragments straight from global (rows are per-lane contiguous)
    v16bf qa[3];
    #pragma unroll
    for (int kc = 0; kc < 3; ++kc)
      qa[kc] = load_frag(Qg, HDIM, rt * 16 + lrow, kc * 32, laneHi);

    // scores S = scale * Q K^T  -> fp32 LDS scratch
    for (int ct = 0; ct < SEQP / 16; ++ct) {
      v8f acc = {};
      #pragma unroll
      for (int kc = 0; kc < 3; ++kc) {
        v16bf kb = load_frag(Ks, KS_STRIDE, ct * 16 + lrow, kc * 32, laneHi);
        acc = wmma_bf16(qa[kc], kb, acc);
      }
      #pragma unroll
      for (int r = 0; r < 8; ++r) {
        int rr = laneHi ? (r + 8) : r;
        Sw[rr * SEQP2 + ct * 16 + lrow] = acc[r] * scale;
      }
    }
    asm volatile("s_wait_dscnt 0" ::: "memory");

    // block edits + softmax: lanes (lrow, lrow+16) share row lrow, split columns
    {
      const int gr = rt * 16 + lrow;
      float* Srow = Sw + lrow * SEQP2;
      if (laneHi) {
        if (gr < S1B) {
          for (int c = S2B; c < SEQ; ++c) Srow[c] = Srow[c - (SEQ - S2B)]; // copy pre-bias
          for (int c = S1B; c < S2B; ++c) Srow[c] -= BIASV;
        } else if (gr < S2B) {
          for (int c = S2B; c < SEQ; ++c) Srow[c] -= BIASV;
        } else {
          for (int c = S1B; c < S2B; ++c) Srow[c] -= BIASV;
        }
        for (int c = SEQ; c < SEQP2; ++c) Srow[c] = -__builtin_inff();
      }
      asm volatile("s_wait_dscnt 0" ::: "memory");

      const int cb = laneHi * 128;
      float m = -__builtin_inff();
      for (int c = 0; c < 128; ++c) m = fmaxf(m, Srow[cb + c]);
      m = fmaxf(m, __shfl_xor(m, 16, 32));
      float s = 0.0f;
      for (int c = 0; c < 128; ++c) {
        float e = __expf(Srow[cb + c] - m);
        Srow[cb + c] = e;
        s += e;
      }
      s += __shfl_xor(s, 16, 32);
      float inv = 1.0f / s;
      __bf16* Prow = Pw + lrow * P_STRIDE;
      for (int c = 0; c < 128; ++c) Prow[cb + c] = (__bf16)(Srow[cb + c] * inv);
    }
    asm volatile("s_wait_dscnt 0" ::: "memory");

    // out tile = P (16 x 256) @ V (256 x 96) via Vt rows (= head dims)
    v8f oacc[HDIM / 16] = {};
    for (int kc = 0; kc < SEQP2 / 32; ++kc) {
      v16bf pa = load_frag(Pw, P_STRIDE, lrow, kc * 32, laneHi);
      #pragma unroll
      for (int ot = 0; ot < HDIM / 16; ++ot) {
        v16bf vb = load_frag(Vt, VT_STRIDE, ot * 16 + lrow, kc * 32, laneHi);
        oacc[ot] = wmma_bf16(pa, vb, oacc[ot]);
      }
    }

    // store context in [B, SEQP, H*D] (bf16), padded rows included (discarded later)
    #pragma unroll
    for (int ot = 0; ot < HDIM / 16; ++ot)
      #pragma unroll
      for (int r = 0; r < 8; ++r) {
        int rr = laneHi ? (r + 8) : r;
        int gr = rt * 16 + rr;
        int gd = ot * 16 + lrow;
        ctx[((size_t)(b * SEQP + gr)) * CH + h * HDIM + gd] = (__bf16)oacc[ot][r];
      }
  }
}

// ---------------------------------------------------------------------------
// Host launch
// ---------------------------------------------------------------------------
extern "C" void kernel_launch(void* const* d_in, const int* in_sizes, int n_in,
                              void* d_out, int out_size, void* d_ws, size_t ws_size,
                              hipStream_t stream) {
  const float* x      = (const float*)d_in[0];   // [64,236,768]
  const float* qkv_w  = (const float*)d_in[1];   // [2304,768]
  const float* proj_w = (const float*)d_in[2];   // [768,768]
  const float* proj_b = (const float*)d_in[3];   // [768]
  float* out = (float*)d_out;                    // [64,236,768]

  char* ws = (char*)d_ws;
  const size_t xh_sz   = (size_t)BATCH * SEQP * CH * 2;            // 23.6 MB
  const size_t wqkv_sz = (size_t)CQKV * CH * 2;                    // 3.5 MB
  const size_t wprj_sz = (size_t)CH * CH * 2;                      // 1.2 MB
  const size_t qkv_sz  = (size_t)3 * BATCH * HEADS * SEQP * HDIM * 2; // 70.8 MB

  __bf16* xh    = (__bf16*)ws;                 ws += xh_sz;
  __bf16* wqkvh = (__bf16*)ws;                 ws += wqkv_sz;
  __bf16* wprjh = (__bf16*)ws;                 ws += wprj_sz;
  __bf16* qkvh  = (__bf16*)ws;                 ws += qkv_sz;
  __bf16* ctx   = (__bf16*)ws;                 // 23.6 MB

  // 1) converts
  cvt_x_kernel<<<(BATCH * SEQP * CH + 255) / 256, 256, 0, stream>>>(x, xh);
  cvt_w_kernel<<<(CQKV * CH + 255) / 256, 256, 0, stream>>>(qkv_w, wqkvh, CQKV * CH);
  cvt_w_kernel<<<(CH * CH + 255) / 256, 256, 0, stream>>>(proj_w, wprjh, CH * CH);

  // 2) QKV GEMM: [15360 x 768] x [2304 x 768]^T
  gemm_kernel<<<dim3((BATCH * SEQP) / 128, CQKV / 128), 256, 0, stream>>>(
      xh, wqkvh, CH, /*mode=*/0, qkvh, nullptr, nullptr);

  // 3) fused attention, one workgroup per (b,h)
  attn_kernel<<<BATCH * HEADS, 256, SMEM_ATTN, stream>>>(qkvh, ctx);

  // 4) proj GEMM + bias: [15360 x 768] x [768 x 768]^T -> fp32 out
  gemm_kernel<<<dim3((BATCH * SEQP) / 128, CH / 128), 256, 0, stream>>>(
      ctx, wprjh, CH, /*mode=*/1, nullptr, out, proj_b);
}
